// UpdateNode_30477087933089
// MI455X (gfx1250) — matrix operand order repaired and checked by reading, not compiled, not run on hardware
//
#include <hip/hip_runtime.h>
#include <math.h>

// ---------------------------------------------------------------------------
// MI455X / gfx1250, wave32. Heavy GEMMs: v_wmma_f32_16x16x32_bf16 with
// register blocking (32x64 per wave) + B-panel staged in LDS via the Tensor
// Data Mover (tensor_load_to_lds + s_wait_tensorcnt).
// Sizes: N=20000 nodes, E=160000 edges, S=128, V=64, LAT=128, OH=32, D=320.
// Workspace requirement: ~1.12 GB (bump-allocated, stage reuse).
// ---------------------------------------------------------------------------

typedef __bf16 bf16_t;
typedef __attribute__((ext_vector_type(16))) __bf16 v16bf;
typedef __attribute__((ext_vector_type(8)))  __bf16 v8bf;
typedef __attribute__((ext_vector_type(8)))  float  v8f;
typedef __attribute__((ext_vector_type(4)))  unsigned int u32x4;
typedef __attribute__((ext_vector_type(8)))  int i32x8;
typedef __attribute__((ext_vector_type(4)))  int i32x4;

#define C_OLD_F 0.894427190999916f            /* 1/sqrt(1.25) */
#define C_AGG_F 0.158113883008419f            /* 0.5*C_OLD/sqrt(8) */

__device__ __forceinline__ float wave_sum32(float x) {
#pragma unroll
  for (int off = 16; off > 0; off >>= 1) x += __shfl_xor(x, off, 32);
  return x;
}

// A fragment (16x32 bf16, M x K): lane r/half holds row r, 16 contiguous K
// values at k0+half*16 (matches B fragment k order)
__device__ __forceinline__ v16bf load_fragA(const bf16_t* __restrict__ A, int ld,
                                            int row0, int k0, int M) {
  const int lane = threadIdx.x & 31;
  const int half = lane >> 4;
  int row = row0 + (lane & 15);
  row = row < M ? row : M - 1;
  const bf16_t* p = A + (size_t)row * ld + k0 + half * 16;
  v8bf lo = *(const v8bf*)(p);
  v8bf hi = *(const v8bf*)(p + 8);
  v16bf f;
#pragma unroll
  for (int i = 0; i < 8; ++i) { f[i] = lo[i]; f[i + 8] = hi[i]; }
  return f;
}

// B fragment (32x16, K x N) from BT (N x K row-major): lane r holds column
// n0+r, 16 contiguous K values at k0+half*16 (same k order as load_fragA)
__device__ __forceinline__ v16bf load_fragB(const bf16_t* __restrict__ BT, int ld,
                                            int n0, int k0) {
  const int lane = threadIdx.x & 31;
  const int half = lane >> 4;
  const bf16_t* p = BT + (size_t)(n0 + (lane & 15)) * ld + k0 + half * 16;
  v8bf lo = *(const v8bf*)(p);
  v8bf hi = *(const v8bf*)(p + 8);
  v16bf f;
#pragma unroll
  for (int i = 0; i < 8; ++i) { f[i] = lo[i]; f[i + 8] = hi[i]; }
  return f;
}

// ---------------------------------------------------------------------------
// TDM: stage a [rows x K] bf16 panel (row-major, stride K) from global to LDS.
// D# built per CDNA5 ISA (2D tensor, groups 2/3 zero).
// ---------------------------------------------------------------------------
__device__ __forceinline__ void tdm_load_panel_bf16(const bf16_t* gsrc,
                                                    unsigned lds_byte,
                                                    int rows, int K) {
  unsigned long long ga = (unsigned long long)(uintptr_t)gsrc;
  u32x4 g0;
  g0[0] = 1u;                                              // count=1
  g0[1] = lds_byte;                                        // lds_addr (bytes)
  g0[2] = (unsigned)(ga & 0xffffffffull);                  // global_addr[31:0]
  g0[3] = (unsigned)((ga >> 32) & 0x1ffffffull) | (2u << 30); // [56:32]|type=2
  i32x8 g1;
  g1[0] = (1 << 16);                                       // data_size=1 (2B)
  g1[1] = (int)(((unsigned)K & 0xffffu) << 16);            // tensor_dim0 lo16
  g1[2] = (int)((((unsigned)K >> 16) & 0xffffu) |          // tensor_dim0 hi16
                (((unsigned)rows & 0xffffu) << 16));       // tensor_dim1 lo16
  g1[3] = (int)((((unsigned)rows >> 16) & 0xffffu) |       // tensor_dim1 hi16
                (((unsigned)K & 0xffffu) << 16));          // tile_dim0 = K
  g1[4] = (int)((unsigned)rows & 0xffffu);                 // tile_dim1, tile_dim2=0
  g1[5] = (int)K;                                          // tensor_dim0_stride
  g1[6] = 0;
  g1[7] = 0;
  i32x4 z4 = {0, 0, 0, 0};
#if defined(__clang_major__) && (__clang_major__ >= 23)
  i32x8 z8 = {0, 0, 0, 0, 0, 0, 0, 0};
  __builtin_amdgcn_tensor_load_to_lds(g0, g1, z4, z4, z8, 0);
#else
  __builtin_amdgcn_tensor_load_to_lds(g0, g1, z4, z4, 0);
#endif
  __builtin_amdgcn_s_wait_tensorcnt(0);
}

// ---------------------------------------------------------------------------
// Register-blocked bf16 GEMM: C(MxN,f32) = A(MxK,bf16) @ B (given as BT NxK).
// Workgroup = 8 waves, each wave computes a 32x64 tile (MB=2 x NB=4).
// The 64-column B panel is staged in LDS once per workgroup via TDM.
// ---------------------------------------------------------------------------
__global__ __launch_bounds__(256) void gemm_bf16_blk(const bf16_t* __restrict__ A,
                                                     const bf16_t* __restrict__ BT,
                                                     const float* __restrict__ bias,
                                                     float* __restrict__ C,
                                                     int M, int N, int K) {
  extern __shared__ bf16_t spanel[];            // 64 x K bf16
  const int wave = threadIdx.x >> 5;
  const int lane = threadIdx.x & 31;
  const int half = lane >> 4;
  const int r = lane & 15;
  const int nblk = blockIdx.y * 64;
  if (wave == 0) {
    tdm_load_panel_bf16(BT + (size_t)nblk * K, (unsigned)(uintptr_t)spanel, 64, K);
  }
  __syncthreads();
  const int m0 = blockIdx.x * 256 + wave * 32;  // MB=2 tiles: rows m0..m0+31
  const v8f vzero = {0.f, 0.f, 0.f, 0.f, 0.f, 0.f, 0.f, 0.f};
  v8f acc[2][4];
#pragma unroll
  for (int mb = 0; mb < 2; ++mb)
#pragma unroll
    for (int nb = 0; nb < 4; ++nb) acc[mb][nb] = vzero;
  for (int k0 = 0; k0 < K; k0 += 32) {
    v16bf a0 = load_fragA(A, K, m0, k0, M);
    v16bf a1 = load_fragA(A, K, m0 + 16, k0, M);
#pragma unroll
    for (int nb = 0; nb < 4; ++nb) {
      const bf16_t* bp = spanel + (size_t)(nb * 16 + r) * K + k0 + half * 16;
      v8bf blo = *(const v8bf*)bp;
      v8bf bhi = *(const v8bf*)(bp + 8);
      v16bf b;
#pragma unroll
      for (int i = 0; i < 8; ++i) { b[i] = blo[i]; b[i + 8] = bhi[i]; }
      acc[0][nb] = __builtin_amdgcn_wmma_f32_16x16x32_bf16(false, a0, false, b,
                                                           (short)0, acc[0][nb],
                                                           false, false);
      acc[1][nb] = __builtin_amdgcn_wmma_f32_16x16x32_bf16(false, a1, false, b,
                                                           (short)0, acc[1][nb],
                                                           false, false);
    }
  }
#pragma unroll
  for (int nb = 0; nb < 4; ++nb) {
    int n = nblk + nb * 16 + r;
    float bb = bias ? bias[n] : 0.f;
#pragma unroll
    for (int mb = 0; mb < 2; ++mb) {
#pragma unroll
      for (int j = 0; j < 8; ++j) {
        int m = m0 + mb * 16 + half * 8 + j;
        if (m < M) C[(size_t)m * N + n] = acc[mb][nb][j] + bb;
      }
    }
  }
}

// ---------------------------------------------------------------------------
// Weight prep: fp32 -> bf16 transposed (BT layout N x K), optional scale/offset
// ---------------------------------------------------------------------------
__global__ void transpose_to_bf16(const float* __restrict__ W, bf16_t* __restrict__ WT,
                                  int K, int N, int ldk, int koff, float scale) {
  int i = blockIdx.x * 256 + threadIdx.x;
  if (i >= K * N) return;
  int k = i / N, n = i - k * N;
  WT[(size_t)n * ldk + koff + k] = (bf16_t)(scale * W[i]);
}

// Wts (S,OH,S)=(128,32,128) -> WTST[k][i][s] (32 x 128 x 128)
__global__ void wts_transpose(const float* __restrict__ W, bf16_t* __restrict__ WT) {
  int i = blockIdx.x * 256 + threadIdx.x;
  if (i >= 128 * 32 * 128) return;
  int s = i >> 12, k = (i >> 7) & 31, ii = i & 127;
  WT[(size_t)k * 16384 + ii * 128 + s] = (bf16_t)W[i];
}
// Wtv (V,OH,V)=(64,32,64) -> WTVT[k][i][v] (32 x 64 x 64)
__global__ void wtv_transpose(const float* __restrict__ W, bf16_t* __restrict__ WT) {
  int i = blockIdx.x * 256 + threadIdx.x;
  if (i >= 64 * 32 * 64) return;
  int v = i >> 11, k = (i >> 6) & 31, ii = i & 63;
  WT[(size_t)k * 4096 + ii * 64 + v] = (bf16_t)W[i];
}

__global__ void lat_to_bf16(const float* __restrict__ lat, const int* __restrict__ act,
                            bf16_t* __restrict__ out, int E) {
  int i = blockIdx.x * 256 + threadIdx.x;
  if (i >= E * 128) return;
  int e = i >> 7, j = i & 127;
  out[i] = (bf16_t)lat[(size_t)act[e] * 128 + j];
}

__global__ void zero_f32(float* __restrict__ p, int n) {
  int i = blockIdx.x * 256 + threadIdx.x;
  if (i < n) p[i] = 0.f;
}

// ---------------------------------------------------------------------------
// Node scalar/vector layernorm (one wave per node)
// ---------------------------------------------------------------------------
__global__ void node_sln(const float* __restrict__ X, const float* __restrict__ g_s,
                         const float* __restrict__ b_s, const float* __restrict__ g_v,
                         float* __restrict__ NS, float* __restrict__ NV, int Nn) {
  int node = blockIdx.x * (blockDim.x >> 5) + (threadIdx.x >> 5);
  if (node >= Nn) return;
  int lane = threadIdx.x & 31;
  const float* x = X + (size_t)node * 320;
  float s[4], acc = 0.f;
#pragma unroll
  for (int t = 0; t < 4; ++t) { s[t] = x[lane + t * 32]; acc += s[t]; }
  float mu = wave_sum32(acc) * (1.f / 128.f);
  float var = 0.f;
#pragma unroll
  for (int t = 0; t < 4; ++t) { float d = s[t] - mu; var += d * d; }
  float rstd = rsqrtf(wave_sum32(var) * (1.f / 128.f) + 1e-8f);
#pragma unroll
  for (int t = 0; t < 4; ++t) {
    int j = lane + t * 32;
    NS[(size_t)node * 128 + j] = (s[t] - mu) * rstd * g_s[j] + b_s[j];
  }
  float v6[6], ss = 0.f;
#pragma unroll
  for (int t = 0; t < 6; ++t) { v6[t] = x[128 + lane + t * 32]; ss += v6[t] * v6[t]; }
  float rn = rsqrtf(wave_sum32(ss) * (1.f / 192.f) + 1e-8f);
#pragma unroll
  for (int t = 0; t < 6; ++t) {
    int idx = lane + t * 32;
    NV[(size_t)node * 192 + idx] = v6[t] * rn * g_v[idx / 3];
  }
}

// ---------------------------------------------------------------------------
// Edge pre: edge SLN + gather + rotation + latent gating -> bf16 GEMM inputs
// ---------------------------------------------------------------------------
__global__ void edge_pre(const float* __restrict__ EF, const float* __restrict__ g_es,
                         const float* __restrict__ b_es, const float* __restrict__ g_ev,
                         const float* __restrict__ NS, const float* __restrict__ NV,
                         const float* __restrict__ EV, const float* __restrict__ G,
                         const int* __restrict__ EIDX, const int* __restrict__ ACT,
                         bf16_t* __restrict__ A0, bf16_t* __restrict__ AXY,
                         float* __restrict__ RW, int E) {
  int e = blockIdx.x * (blockDim.x >> 5) + (threadIdx.x >> 5);
  if (e >= E) return;
  int lane = threadIdx.x & 31;
  int ae = ACT[e];
  int ctr = EIDX[ae];
  int ngh = EIDX[E + ae];
  const float* x = EF + (size_t)e * 320;
  float s4[4], acc = 0.f;
#pragma unroll
  for (int t = 0; t < 4; ++t) { s4[t] = x[lane + t * 32]; acc += s4[t]; }
  float mu = wave_sum32(acc) * (1.f / 128.f);
  float var = 0.f;
#pragma unroll
  for (int t = 0; t < 4; ++t) { float d = s4[t] - mu; var += d * d; }
  float rstd = rsqrtf(wave_sum32(var) * (1.f / 128.f) + 1e-8f);
  float ss = 0.f;
#pragma unroll
  for (int t = 0; t < 6; ++t) { float v = x[128 + lane + t * 32]; ss += v * v; }
  float rsn = rsqrtf(wave_sum32(ss) * (1.f / 192.f) + 1e-8f);
  float e0 = EV[(size_t)ae * 3 + 0], e1 = EV[(size_t)ae * 3 + 1], e2 = EV[(size_t)ae * 3 + 2];
  float rv = rsqrtf(e0 * e0 + e1 * e1 + e2 * e2 + 1e-12f);
  float nx = e0 * rv, ny = e1 * rv, nz = e2 * rv;
  float fx = 1.f / (1.f + nz + 1e-8f);
  float R00 = 1.f - nx * nx * fx, R01 = -nx * ny * fx, R02 = -nx;
  float R10 = R01, R11 = 1.f - ny * ny * fx, R12 = -ny;
  float R20 = nx, R21 = ny, R22 = 1.f - (nx * nx + ny * ny) * fx;
  if (lane == 0) {
    float* rw = RW + (size_t)e * 9;
    rw[0] = R00; rw[1] = R01; rw[2] = R02;
    rw[3] = R10; rw[4] = R11; rw[5] = R12;
    rw[6] = R20; rw[7] = R21; rw[8] = R22;
  }
  const float* grow = G + (size_t)e * 576;
  bf16_t* a0 = A0 + (size_t)e * 576;
  bf16_t* axy = AXY + (size_t)e * 384;
#pragma unroll
  for (int t = 0; t < 12; ++t) {
    int j = lane + t * 32;
    float sval;
    if (j < 128)       sval = NS[(size_t)ctr * 128 + j];
    else if (j < 256)  { int q = j - 128; sval = (s4[t - 4] - mu) * rstd * g_es[q] + b_es[q]; }
    else               sval = NS[(size_t)ngh * 128 + (j - 256)];
    a0[j] = (bf16_t)(sval * grow[j]);
  }
#pragma unroll
  for (int t = 0; t < 6; ++t) {
    int vv = lane + t * 32;
    float vx_, vy_, vz_;
    if (vv < 64) {
      const float* p = NV + (size_t)ctr * 192 + vv * 3;
      vx_ = p[0]; vy_ = p[1]; vz_ = p[2];
    } else if (vv < 128) {
      int q = vv - 64; const float* p = x + 128 + q * 3;
      float sc = rsn * g_ev[q];
      vx_ = p[0] * sc; vy_ = p[1] * sc; vz_ = p[2] * sc;
    } else {
      const float* p = NV + (size_t)ngh * 192 + (vv - 128) * 3;
      vx_ = p[0]; vy_ = p[1]; vz_ = p[2];
    }
    float rx = R00 * vx_ + R01 * vy_ + R02 * vz_;
    float ry = R10 * vx_ + R11 * vy_ + R12 * vz_;
    float rz = R20 * vx_ + R21 * vy_ + R22 * vz_;
    float gg = grow[384 + vv];
    a0[384 + vv]   = (bf16_t)(rz * gg);
    axy[vv]        = (bf16_t)(rx * gg);
    axy[192 + vv]  = (bf16_t)(ry * gg);
  }
}

// ---------------------------------------------------------------------------
// Edge post: v_out = R^T @ [ox,oy,vz], silu/sigmoid gates -> bf16 inputs
// ---------------------------------------------------------------------------
__global__ void edge_post(const float* __restrict__ M0v, const float* __restrict__ OX,
                          const float* __restrict__ OY, const float* __restrict__ RW,
                          bf16_t* __restrict__ AS, bf16_t* __restrict__ AV, int E) {
  int e = blockIdx.x * (blockDim.x >> 5) + (threadIdx.x >> 5);
  if (e >= E) return;
  int lane = threadIdx.x & 31;
  const float* m0row = M0v + (size_t)e * 256;
  float R[9];
#pragma unroll
  for (int q = 0; q < 9; ++q) R[q] = RW[(size_t)e * 9 + q];
#pragma unroll
  for (int t = 0; t < 4; ++t) {
    int j = lane + t * 32;
    float so = m0row[j];
    AS[(size_t)e * 128 + j] = (bf16_t)(so / (1.f + expf(-so)));  // silu
  }
#pragma unroll
  for (int t = 0; t < 2; ++t) {
    int vv = lane + t * 32;
    float gate = 1.f / (1.f + expf(-m0row[128 + vv]));
    float ax = OX[(size_t)e * 64 + vv];
    float ay = OY[(size_t)e * 64 + vv];
    float az = m0row[192 + vv];
    float ox_ = R[0] * ax + R[3] * ay + R[6] * az;  // R^T
    float oy_ = R[1] * ax + R[4] * ay + R[7] * az;
    float oz_ = R[2] * ax + R[5] * ay + R[8] * az;
    AV[((size_t)0 * E + e) * 64 + vv] = (bf16_t)(ox_ * gate);
    AV[((size_t)1 * E + e) * 64 + vv] = (bf16_t)(oy_ * gate);
    AV[((size_t)2 * E + e) * 64 + vv] = (bf16_t)(oz_ * gate);
  }
}

// ---------------------------------------------------------------------------
// Edge messages out + weighted segment-sum (atomic) into AGG
// ---------------------------------------------------------------------------
__global__ void edge_msg(const float* __restrict__ SMSG, const float* __restrict__ VMSG,
                         const float* __restrict__ WOUT, const int* __restrict__ EIDX,
                         const int* __restrict__ ACT,
                         float* __restrict__ EM, float* __restrict__ AGG, int E) {
  int e = blockIdx.x * (blockDim.x >> 5) + (threadIdx.x >> 5);
  if (e >= E) return;
  int lane = threadIdx.x & 31;
  int ctr = EIDX[ACT[e]];
  const float* sm = SMSG + (size_t)e * 128;
  const float* wrow = WOUT + (size_t)e * 192;
  float* em = EM + (size_t)e * 320;
  float* ag = AGG + (size_t)ctr * 320;
#pragma unroll
  for (int t = 0; t < 4; ++t) {
    int j = lane + t * 32;
    float sv = sm[j];
    em[j] = sv;
    atomicAdd(&ag[j], sv * wrow[j]);
  }
#pragma unroll
  for (int t = 0; t < 2; ++t) {
    int vv = lane + t * 32;
    float wv = wrow[128 + vv];
#pragma unroll
    for (int c = 0; c < 3; ++c) {
      float mv = VMSG[((size_t)c * E + e) * 64 + vv];
      em[128 + vv * 3 + c] = mv;
      atomicAdd(&ag[128 + vv * 3 + c], mv * wv);
    }
  }
}

// ---------------------------------------------------------------------------
// nf = C_OLD * node_features + C_AGG * agg ; also emit bf16 splits for Wts/Wtv
// ---------------------------------------------------------------------------
__global__ void node_final(const float* __restrict__ NFT, const float* __restrict__ AGG,
                           float* __restrict__ NFOUT, bf16_t* __restrict__ SFB,
                           bf16_t* __restrict__ VFB, int Nn) {
  int i = blockIdx.x * 256 + threadIdx.x;
  if (i >= Nn * 320) return;
  float v = C_OLD_F * NFT[i] + C_AGG_F * AGG[i];
  NFOUT[i] = v;
  int n = i / 320, j = i - n * 320;
  if (j < 128) {
    SFB[(size_t)n * 128 + j] = (bf16_t)v;
  } else {
    int q = j - 128, vv = q / 3, c = q - vv * 3;
    VFB[((size_t)c * Nn + n) * 64 + vv] = (bf16_t)v;
  }
}

// ---------------------------------------------------------------------------
// ts[n,i] = sum_k oh[n,k] * (s_f[n,:] @ Wts[:,k,:])[i]  — per-k WMMA with
// row-scaled f32 accumulation; += into nf scalar part.
// ---------------------------------------------------------------------------
__global__ __launch_bounds__(32) void ts_kernel(const bf16_t* __restrict__ SF,
                                                const bf16_t* __restrict__ WTST,
                                                const float* __restrict__ OH,
                                                float* __restrict__ NF, int Nn) {
  const int n0 = blockIdx.x * 16;
  const int i0 = blockIdx.y * 16;
  const int lane = threadIdx.x & 31;
  const int half = lane >> 4;
  const int r = lane & 15;
  v16bf a[4];
#pragma unroll
  for (int kk = 0; kk < 4; ++kk) a[kk] = load_fragA(SF, 128, n0, kk * 32, Nn);
  v8f acc = {0.f, 0.f, 0.f, 0.f, 0.f, 0.f, 0.f, 0.f};
  for (int k = 0; k < 32; ++k) {
    const bf16_t* B = WTST + (size_t)k * 16384;
    v8f d = {0.f, 0.f, 0.f, 0.f, 0.f, 0.f, 0.f, 0.f};
#pragma unroll
    for (int kk = 0; kk < 4; ++kk) {
      v16bf b = load_fragB(B, 128, i0, kk * 32);
      d = __builtin_amdgcn_wmma_f32_16x16x32_bf16(false, a[kk], false, b, (short)0,
                                                  d, false, false);
    }
#pragma unroll
    for (int j = 0; j < 8; ++j)
      acc[j] += OH[(size_t)(n0 + half * 8 + j) * 32 + k] * d[j];
  }
#pragma unroll
  for (int j = 0; j < 8; ++j) {
    int n = n0 + half * 8 + j;
    if (n < Nn) NF[(size_t)n * 320 + (i0 + r)] += acc[j];
  }
}

// ---------------------------------------------------------------------------
// tv[n,i,c]: rows are (c,n) over channel-major vf planes; += into nf vec part.
// ---------------------------------------------------------------------------
__global__ __launch_bounds__(32) void tv_kernel(const bf16_t* __restrict__ VF,
                                                const bf16_t* __restrict__ WTVT,
                                                const float* __restrict__ OH,
                                                float* __restrict__ NF, int Nn) {
  const int row0 = blockIdx.x * 16;
  const int c = row0 / Nn;
  const int n0 = row0 - c * Nn;
  const int i0 = blockIdx.y * 16;
  const int lane = threadIdx.x & 31;
  const int half = lane >> 4;
  const int r = lane & 15;
  const bf16_t* Ap = VF + (size_t)c * Nn * 64;
  v16bf a[2];
#pragma unroll
  for (int kk = 0; kk < 2; ++kk) a[kk] = load_fragA(Ap, 64, n0, kk * 32, Nn);
  v8f acc = {0.f, 0.f, 0.f, 0.f, 0.f, 0.f, 0.f, 0.f};
  for (int k = 0; k < 32; ++k) {
    const bf16_t* B = WTVT + (size_t)k * 4096;
    v8f d = {0.f, 0.f, 0.f, 0.f, 0.f, 0.f, 0.f, 0.f};
#pragma unroll
    for (int kk = 0; kk < 2; ++kk) {
      v16bf b = load_fragB(B, 64, i0, kk * 32);
      d = __builtin_amdgcn_wmma_f32_16x16x32_bf16(false, a[kk], false, b, (short)0,
                                                  d, false, false);
    }
#pragma unroll
    for (int j = 0; j < 8; ++j)
      acc[j] += OH[(size_t)(n0 + half * 8 + j) * 32 + k] * d[j];
  }
#pragma unroll
  for (int j = 0; j < 8; ++j) {
    int n = n0 + half * 8 + j;
    if (n < Nn) NF[(size_t)n * 320 + 128 + 3 * (i0 + r) + c] += acc[j];
  }
}

// ---------------------------------------------------------------------------
extern "C" void kernel_launch(void* const* d_in, const int* in_sizes, int n_in,
                              void* d_out, int out_size, void* d_ws, size_t ws_size,
                              hipStream_t stream) {
  (void)n_in; (void)out_size; (void)ws_size;
  const float* latents       = (const float*)d_in[0];
  const float* node_features = (const float*)d_in[1];
  const float* edge_features = (const float*)d_in[2];
  const float* node_onehot   = (const float*)d_in[3];
  const float* edge_vector   = (const float*)d_in[4];
  const float* g_ns = (const float*)d_in[5];
  const float* b_ns = (const float*)d_in[6];
  const float* g_nv = (const float*)d_in[7];
  const float* g_es = (const float*)d_in[8];
  const float* b_es = (const float*)d_in[9];
  const float* g_ev = (const float*)d_in[10];
  const float* Wg  = (const float*)d_in[11];
  const float* bg  = (const float*)d_in[12];
  const float* W0  = (const float*)d_in[13];
  const float* Wr  = (const float*)d_in[14];
  const float* Wi  = (const float*)d_in[15];
  const float* Wps = (const float*)d_in[16];
  const float* bps = (const float*)d_in[17];
  const float* Wpv = (const float*)d_in[18];
  const float* We  = (const float*)d_in[19];
  const float* be  = (const float*)d_in[20];
  const float* Wts = (const float*)d_in[21];
  const float* Wtv = (const float*)d_in[22];
  const int* edge_index = (const int*)d_in[24];
  const int* active     = (const int*)d_in[25];

  const int Nn = in_sizes[1] / 320;
  const int E  = in_sizes[2] / 320;

  size_t off = 0;
  auto take = [&](size_t bytes) -> char* {
    char* p = (char*)d_ws + off;
    off += (bytes + 255) & ~(size_t)255;
    return p;
  };
  bf16_t* WGT  = (bf16_t*)take((size_t)576 * 128 * 2);
  bf16_t* W0T  = (bf16_t*)take((size_t)256 * 576 * 2);
  bf16_t* WROT = (bf16_t*)take((size_t)64 * 384 * 2);
  bf16_t* WIOT = (bf16_t*)take((size_t)64 * 384 * 2);
  bf16_t* WPST = (bf16_t*)take((size_t)128 * 128 * 2);
  bf16_t* WPVT = (bf16_t*)take((size_t)64 * 64 * 2);
  bf16_t* WET  = (bf16_t*)take((size_t)192 * 128 * 2);
  bf16_t* WTST = (bf16_t*)take((size_t)32 * 128 * 128 * 2);
  bf16_t* WTVT = (bf16_t*)take((size_t)32 * 64 * 64 * 2);
  bf16_t* LATB = (bf16_t*)take((size_t)E * 128 * 2);
  float*  NS   = (float*)take((size_t)Nn * 128 * 4);
  float*  NV   = (float*)take((size_t)Nn * 192 * 4);
  char*   big  = take((size_t)E * 576 * 4);   // G, later reused for M0|OX|OY
  float*  G    = (float*)big;
  float*  M0f  = (float*)big;
  float*  OX   = (float*)(big + (size_t)E * 256 * 4);
  float*  OY   = (float*)(big + (size_t)E * 256 * 4 + (size_t)E * 64 * 4);
  char*   big2 = take((size_t)E * 576 * 2);   // A0, later reused for AS|AV
  bf16_t* A0   = (bf16_t*)big2;
  bf16_t* AS   = (bf16_t*)big2;
  bf16_t* AV   = (bf16_t*)(big2 + (size_t)E * 128 * 2);
  bf16_t* AXY  = (bf16_t*)take((size_t)E * 384 * 2);
  float*  RW   = (float*)take((size_t)E * 9 * 4);
  float*  SMSG = (float*)take((size_t)E * 128 * 4);
  float*  VMSG = (float*)take((size_t)E * 192 * 4);
  float*  WOUT = (float*)take((size_t)E * 192 * 4);
  float*  AGG  = (float*)take((size_t)Nn * 320 * 4);
  bf16_t* SFB  = (bf16_t*)take((size_t)Nn * 128 * 2);
  bf16_t* VFB  = (bf16_t*)take((size_t)Nn * 192 * 2);

  float* nf_out = (float*)d_out;
  float* em_out = nf_out + (size_t)Nn * 320;

  auto tc = [&](const float* W, bf16_t* WT, int K, int N, int ldk, int koff, float sc) {
    int total = K * N;
    transpose_to_bf16<<<(total + 255) / 256, 256, 0, stream>>>(W, WT, K, N, ldk, koff, sc);
  };
  tc(Wg,  WGT, 128, 576, 128, 0, 1.f);
  tc(W0,  W0T, 576, 256, 576, 0, 1.f);
  tc(Wr,  WROT, 192, 64, 384, 0,  1.f);   // ox = vx@Wr - vy@Wi
  tc(Wi,  WROT, 192, 64, 384, 192, -1.f);
  tc(Wi,  WIOT, 192, 64, 384, 0,  1.f);   // oy = vx@Wi + vy@Wr
  tc(Wr,  WIOT, 192, 64, 384, 192, 1.f);
  tc(Wps, WPST, 128, 128, 128, 0, 1.f);
  tc(Wpv, WPVT, 64, 64, 64, 0, 1.f);
  tc(We,  WET, 128, 192, 128, 0, 1.f);
  wts_transpose<<<(128 * 32 * 128 + 255) / 256, 256, 0, stream>>>(Wts, WTST);
  wtv_transpose<<<(64 * 32 * 64 + 255) / 256, 256, 0, stream>>>(Wtv, WTVT);
  lat_to_bf16<<<(E * 128 + 255) / 256, 256, 0, stream>>>(latents, active, LATB, E);

  node_sln<<<(Nn + 7) / 8, 256, 0, stream>>>(node_features, g_ns, b_ns, g_nv, NS, NV, Nn);

  // blocked WMMA GEMM: grid = (M/256, N/64), 8 waves/WG, TDM-staged B panel
  auto gemm = [&](const bf16_t* A, const bf16_t* BT, const float* bias, float* C,
                  int M, int Ncol, int K) {
    dim3 g((M + 255) / 256, Ncol / 64);
    size_t sh = (size_t)64 * K * sizeof(bf16_t);
    gemm_bf16_blk<<<g, 256, sh, stream>>>(A, BT, bias, C, M, Ncol, K);
  };

  gemm(LATB, WGT, bg, G, E, 576, 128);                 // g = lat@Wg + bg
  edge_pre<<<(E + 7) / 8, 256, 0, stream>>>(edge_features, g_es, b_es, g_ev,
                                            NS, NV, edge_vector, G, edge_index,
                                            active, A0, AXY, RW, E);
  gemm(A0,  W0T,  nullptr, M0f, E, 256, 576);          // m0
  gemm(AXY, WROT, nullptr, OX,  E, 64, 384);           // ox
  gemm(AXY, WIOT, nullptr, OY,  E, 64, 384);           // oy
  edge_post<<<(E + 7) / 8, 256, 0, stream>>>(M0f, OX, OY, RW, AS, AV, E);
  gemm(AS,   WPST, bps, SMSG, E, 128, 128);            // s_msg
  gemm(AV,   WPVT, nullptr, VMSG, 3 * E, 64, 64);      // v_msg (3 planes as rows)
  gemm(LATB, WET,  be, WOUT, E, 192, 128);             // w
  zero_f32<<<(Nn * 320 + 255) / 256, 256, 0, stream>>>(AGG, Nn * 320);
  edge_msg<<<(E + 7) / 8, 256, 0, stream>>>(SMSG, VMSG, WOUT, edge_index, active,
                                            em_out, AGG, E);
  node_final<<<(Nn * 320 + 255) / 256, 256, 0, stream>>>(node_features, AGG,
                                                         nf_out, SFB, VFB, Nn);
  ts_kernel<<<dim3(Nn / 16, 8), 32, 0, stream>>>(SFB, WTST, node_onehot, nf_out, Nn);
  tv_kernel<<<dim3(3 * Nn / 16, 4), 32, 0, stream>>>(VFB, WTVT, node_onehot, nf_out, Nn);
}
